// FFA_RGNet_32014686224806
// MI455X (gfx1250) — matrix operand order repaired and checked by reading, not compiled, not run on hardware
//
#include <hip/hip_runtime.h>

#define B_    64
#define T_    16
#define D_    2048
#define DFF_  2048
#define H_    8
#define L_    2
#define E_    32
#define HID_  64
#define GOUT_ 32
#define S_    (B_*T_)
#define DH_   (D_/H_)
#define N9_   (S_*9)
#define NEPAT_ (S_*58)
#define NE_   (NEPAT_ + 2*S_)
#define THRESH_ 0.05f

typedef unsigned int u32;
typedef unsigned short u16;
typedef __attribute__((ext_vector_type(4)))  u32    u32x4;
typedef __attribute__((ext_vector_type(16))) __bf16 v16bf;
typedef __attribute__((ext_vector_type(8)))  float  v8f;

union Frag16 { v16bf v; u32x4 q[2]; };

__device__ __forceinline__ u16 f2bf(float f) {
    u32 u = __float_as_uint(f);
    u32 r = u + 0x7FFFu + ((u >> 16) & 1u);   // round-to-nearest-even
    return (u16)(r >> 16);
}

// LDS byte offset of a shared-memory pointer (generic LDS ptr: low 32 bits = offset)
__device__ __forceinline__ u32 lds_off(const void* p) {
    return (u32)(size_t)p;
}

// CDNA5 async DMA: 16B per lane, global -> LDS, tracked by ASYNCcnt
__device__ __forceinline__ void async_copy_b128(u32 dst_lds, const void* src_global) {
    asm volatile("global_load_async_to_lds_b128 %0, %1, off"
                 :: "v"(dst_lds), "v"(src_global) : "memory");
}

__device__ __forceinline__ float block_reduce_sum(float v, float* sh) {
    int tid = threadIdx.x;
    sh[tid] = v;
    __syncthreads();
    #pragma unroll
    for (int off = 128; off > 0; off >>= 1) {
        if (tid < off) sh[tid] += sh[tid + off];
        __syncthreads();
    }
    float r = sh[0];
    __syncthreads();
    return r;
}

// ---------------- elementwise helpers ----------------
__global__ void k_f32_to_bf16(const float* __restrict__ in, u16* __restrict__ out, int n) {
    int i = blockIdx.x * blockDim.x + threadIdx.x;
    if (i < n) out[i] = f2bf(in[i]);
}
__global__ void k_copy(const float* __restrict__ a, float* __restrict__ b, int n) {
    int i = blockIdx.x * blockDim.x + threadIdx.x;
    if (i < n) b[i] = a[i];
}
__global__ void k_fill(float* __restrict__ p, float val, int n) {
    int i = blockIdx.x * blockDim.x + threadIdx.x;
    if (i < n) p[i] = val;
}

// ---------------- WMMA bf16 GEMM: C[M,N] = A[M,K] @ W[N,K]^T (+bias, relu) ---
// Block = 128 threads (4 waves, 2x2 wave grid), block tile 64x128, wave tile 32x64.
// K staged in chunks of 64 via async DMA into double-buffered LDS.
// LDS row stride 72 u16 (144B = 36 banks): 16 consecutive rows hit distinct bank
// groups for ds_load_b128, and 144 % 16 == 0 keeps 16B alignment.
#define LDA_ 72

__global__ __launch_bounds__(128) void k_gemm(const u16* __restrict__ A,
                                              const u16* __restrict__ Wt,
                                              const float* __restrict__ bias,
                                              float* __restrict__ C,
                                              int M, int N, int K, int relu) {
    __shared__ __align__(16) u16 smA[2][64][LDA_];    // 64 rows  x 64 K
    __shared__ __align__(16) u16 smB[2][128][LDA_];   // 128 rows x 64 K

    int tid  = threadIdx.x;
    int lane = tid & 31;
    int wave = tid >> 5;
    int wM = wave >> 1, wN = wave & 1;
    int n0 = blockIdx.x * 128;
    int m0 = blockIdx.y * 64;

    int ml  = lane & 15;
    int akb = (lane >> 4) * 8;    // A frag: lanes 16-31 carry K+8 halves
    int bkb = (lane >> 4) * 8;    // B frag: lanes 16-31 carry K+8 halves

    // Stage one K-chunk (64): A tile 512 16B-chunks + B tile 1024 16B-chunks
    // over 128 threads -> 12 async ops per thread per stage.
    auto issue = [&](int buf, int kk) {
        #pragma unroll
        for (int r = 0; r < 4; ++r) {
            int c = tid + r * 128;
            int row = c >> 3, part = (c & 7) * 8;
            async_copy_b128(lds_off(&smA[buf][row][part]),
                            A + (size_t)(m0 + row) * K + kk + part);
        }
        #pragma unroll
        for (int r = 0; r < 8; ++r) {
            int c = tid + r * 128;
            int row = c >> 3, part = (c & 7) * 8;
            async_copy_b128(lds_off(&smB[buf][row][part]),
                            Wt + (size_t)(n0 + row) * K + kk + part);
        }
    };

    v8f acc[2][4] = {};

    auto compute = [&](int buf) {
        #pragma unroll
        for (int ks = 0; ks < 64; ks += 32) {
            Frag16 fa[2], fb[4];
            #pragma unroll
            for (int u = 0; u < 2; ++u) {
                const u16* p = &smA[buf][wM * 32 + u * 16 + ml][ks];
                fa[u].q[0] = *reinterpret_cast<const u32x4*>(p + akb);
                fa[u].q[1] = *reinterpret_cast<const u32x4*>(p + akb + 16);
            }
            #pragma unroll
            for (int t = 0; t < 4; ++t) {
                const u16* p = &smB[buf][wN * 64 + t * 16 + ml][ks];
                fb[t].q[0] = *reinterpret_cast<const u32x4*>(p + bkb);
                fb[t].q[1] = *reinterpret_cast<const u32x4*>(p + bkb + 16);
            }
            #pragma unroll
            for (int u = 0; u < 2; ++u)
                #pragma unroll
                for (int t = 0; t < 4; ++t)
                    acc[u][t] = __builtin_amdgcn_wmma_f32_16x16x32_bf16(
                        false, fa[u].v, false, fb[t].v, (short)0, acc[u][t], false, false);
        }
    };

    int nk = K / 64;
    issue(0, 0);
    // Steady state: branch-free issue(next) -> wait(cur) -> compute(cur)
    for (int i = 0; i < nk - 1; ++i) {
        int cur = i & 1;
        issue(1 - cur, (i + 1) * 64);
        asm volatile("s_wait_asynccnt 0xc" ::: "memory");   // cur stage landed
        __syncthreads();
        compute(cur);
        __syncthreads();
    }
    asm volatile("s_wait_asynccnt 0x0" ::: "memory");
    __syncthreads();
    compute((nk - 1) & 1);

    #pragma unroll
    for (int u = 0; u < 2; ++u) {
        #pragma unroll
        for (int t = 0; t < 4; ++t) {
            int n = n0 + wN * 64 + t * 16 + ml;
            float bv = bias ? bias[n] : 0.0f;
            #pragma unroll
            for (int r = 0; r < 8; ++r) {
                int m = m0 + wM * 32 + u * 16 + r + (lane >> 4) * 8;
                float val = acc[u][t][r] + bv;
                if (relu) val = fmaxf(val, 0.0f);
                C[(size_t)m * N + n] = val;
            }
        }
    }
}

// ---------------- heads: phase/loc/les logits + ids ----------------
__global__ __launch_bounds__(256) void k_heads(const float* __restrict__ feat,
        const float* __restrict__ wph, const float* __restrict__ bph,
        const float* __restrict__ wlo, const float* __restrict__ blo,
        const float* __restrict__ wle, const float* __restrict__ ble,
        float* __restrict__ outP, float* __restrict__ outL, float* __restrict__ outE,
        int* __restrict__ pid, int* __restrict__ lid, int* __restrict__ lact) {
    __shared__ float sh[256];
    int s = blockIdx.x, tid = threadIdx.x;
    float pp[3] = {}, pl[7] = {}, pe[7] = {};
    for (int k = tid; k < D_; k += 256) {
        float f = feat[(size_t)s * D_ + k];
        #pragma unroll
        for (int j = 0; j < 3; ++j) pp[j] += f * wph[k * 3 + j];
        #pragma unroll
        for (int j = 0; j < 7; ++j) pl[j] += f * wlo[k * 7 + j];
        #pragma unroll
        for (int j = 0; j < 7; ++j) pe[j] += f * wle[k * 7 + j];
    }
    float rp[3], rl[7], re[7];
    for (int j = 0; j < 3; ++j) rp[j] = block_reduce_sum(pp[j], sh) + bph[j];
    for (int j = 0; j < 7; ++j) rl[j] = block_reduce_sum(pl[j], sh) + blo[j];
    for (int j = 0; j < 7; ++j) re[j] = block_reduce_sum(pe[j], sh) + ble[j];
    if (tid == 0) {
        int ap = 0, al = 0;
        for (int j = 0; j < 3; ++j) { outP[s * 3 + j] = rp[j]; if (rp[j] > rp[ap]) ap = j; }
        for (int j = 0; j < 7; ++j) { outL[s * 7 + j] = rl[j]; if (rl[j] > rl[al]) al = j; }
        for (int j = 0; j < 7; ++j) { outE[s * 7 + j] = re[j]; lact[s * 7 + j] = (re[j] > 0.0f) ? 1 : 0; }
        pid[s] = ap; lid[s] = al;
    }
}

// ---------------- attention (one block per (b,h)) ----------------
__global__ __launch_bounds__(256) void k_attn(const float* __restrict__ qkv,
                                              const int* __restrict__ mask,
                                              float* __restrict__ ctx) {
    __shared__ float sS[T_][T_];
    __shared__ float sA[T_][T_];
    int bh = blockIdx.x, b = bh / H_, h = bh % H_;
    int tid = threadIdx.x;
    int i = tid >> 4, j = tid & 15;
    size_t base = (size_t)b * T_ * 3 * D_;
    const float* qrow = qkv + base + (size_t)i * 3 * D_ + h * DH_;
    const float* krow = qkv + base + (size_t)j * 3 * D_ + D_ + h * DH_;
    float a = 0.0f;
    for (int d = 0; d < DH_; ++d) a += qrow[d] * krow[d];
    a *= (1.0f / 16.0f);                       // 1/sqrt(Dh), Dh=256
    if (!mask[b * T_ + j]) a = -1e9f;
    sS[i][j] = a;
    __syncthreads();
    float mx = -1e30f;
    for (int jj = 0; jj < T_; ++jj) mx = fmaxf(mx, sS[i][jj]);
    float sum = 0.0f;
    for (int jj = 0; jj < T_; ++jj) sum += expf(sS[i][jj] - mx);
    sA[i][j] = expf(sS[i][j] - mx) / sum;
    __syncthreads();
    int d = tid;                               // DH_ == 256 == blockDim
    for (int ii = 0; ii < T_; ++ii) {
        float acc = 0.0f;
        for (int jj = 0; jj < T_; ++jj)
            acc += sA[ii][jj] * qkv[base + (size_t)jj * 3 * D_ + 2 * D_ + h * DH_ + d];
        ctx[(size_t)(b * T_ + ii) * D_ + h * DH_ + d] = acc;
    }
}

// ---------------- residual + layernorm (one block per row) ----------------
__global__ __launch_bounds__(256) void k_resln(const float* __restrict__ xin,
        const float* __restrict__ add, const float* __restrict__ g,
        const float* __restrict__ bb, float* __restrict__ xout) {
    __shared__ float s1[256], s2[256];
    __shared__ float s_mean, s_rstd;
    int s = blockIdx.x, tid = threadIdx.x;
    float v[D_ / 256];
    float sum = 0.0f, sq = 0.0f;
    #pragma unroll
    for (int r = 0; r < D_ / 256; ++r) {
        int d = tid + r * 256;
        float val = xin[(size_t)s * D_ + d] + add[(size_t)s * D_ + d];
        v[r] = val; sum += val; sq += val * val;
    }
    s1[tid] = sum; s2[tid] = sq;
    __syncthreads();
    for (int off = 128; off > 0; off >>= 1) {
        if (tid < off) { s1[tid] += s1[tid + off]; s2[tid] += s2[tid + off]; }
        __syncthreads();
    }
    if (tid == 0) {
        float m = s1[0] / D_;
        float var = s2[0] / D_ - m * m;
        s_mean = m; s_rstd = rsqrtf(var + 1e-5f);
    }
    __syncthreads();
    #pragma unroll
    for (int r = 0; r < D_ / 256; ++r) {
        int d = tid + r * 256;
        xout[(size_t)s * D_ + d] = (v[r] - s_mean) * s_rstd * g[d] + bb[d];
    }
}

// ---------------- scalar score head ----------------
__global__ __launch_bounds__(256) void k_score(const float* __restrict__ x,
        const float* __restrict__ w, const float* __restrict__ b0,
        float* __restrict__ scores) {
    __shared__ float sh[256];
    int s = blockIdx.x, tid = threadIdx.x;
    float p = 0.0f;
    for (int k = tid; k < D_; k += 256) p += x[(size_t)s * D_ + k] * w[k];
    float r = block_reduce_sum(p, sh);
    if (tid == 0) scores[s] = r + b0[0];
}

// ---------------- masked softmax over T, valid flags ----------------
__global__ void k_wts(const float* __restrict__ scores, const int* __restrict__ mask,
                      float* __restrict__ wts, int* __restrict__ valid) {
    int b = threadIdx.x;
    if (b >= B_) return;
    float sc[T_]; float mx = -1e30f;
    for (int t = 0; t < T_; ++t) {
        sc[t] = mask[b * T_ + t] ? scores[b * T_ + t] : -1e30f;
        mx = fmaxf(mx, sc[t]);
    }
    float sum = 0.0f;
    for (int t = 0; t < T_; ++t) { sc[t] = expf(sc[t] - mx); sum += sc[t]; }
    for (int t = 0; t < T_; ++t) {
        float w = sc[t] / sum;
        wts[b * T_ + t] = w;
        valid[b * T_ + t] = (mask[b * T_ + t] != 0 && w >= THRESH_) ? 1 : 0;
    }
}

__global__ __launch_bounds__(256) void k_pooled(const float* __restrict__ x,
        const float* __restrict__ wts, float* __restrict__ pooled) {
    int b = blockIdx.x;
    for (int d = threadIdx.x; d < D_; d += 256) {
        float a = 0.0f;
        for (int t = 0; t < T_; ++t)
            a += wts[b * T_ + t] * x[(size_t)(b * T_ + t) * D_ + d];
        pooled[(size_t)b * D_ + d] = a;
    }
}

// ---------------- reverse scan for next-valid ----------------
__global__ void k_scan(const int* __restrict__ valid, int* __restrict__ nxt) {
    if (threadIdx.x == 0 && blockIdx.x == 0) {
        int carry = -1;
        for (int s = S_ - 1; s >= 0; --s) { nxt[s] = carry; if (valid[s]) carry = s; }
    }
}

// ---------------- graph construction ----------------
__device__ __constant__ int c_srcl[58] = {0,1,1,2,1,3,1,4,1,5,1,6,1,7,1,8,2,3,2,4,2,5,2,6,2,7,2,8,3,4,3,5,3,6,3,7,3,8,4,5,4,6,4,7,4,8,5,6,5,7,5,8,6,7,6,8,7,8};
__device__ __constant__ int c_dstl[58] = {1,0,2,1,3,1,4,1,5,1,6,1,7,1,8,1,3,2,4,2,5,2,6,2,7,2,8,2,4,3,5,3,6,3,7,3,8,3,5,4,6,4,7,4,8,4,6,5,7,5,8,5,7,6,8,6,8,7};

__global__ void k_nodes(const float* __restrict__ embP, const float* __restrict__ embL,
        const float* __restrict__ lew, const float* __restrict__ leb,
        const int* __restrict__ pid, const int* __restrict__ lid,
        const int* __restrict__ lact, const int* __restrict__ valid,
        float* __restrict__ nodes, float* __restrict__ nm) {
    int idx = blockIdx.x * blockDim.x + threadIdx.x;
    if (idx >= N9_) return;
    int s = idx / 9, j = idx % 9;
    float m;
    if (j == 0) {
        const float* e = embP + pid[s] * E_;
        for (int q = 0; q < E_; ++q) nodes[(size_t)idx * E_ + q] = e[q];
        m = valid[s] ? 1.0f : 0.0f;
    } else if (j == 1) {
        const float* e = embL + lid[s] * E_;
        for (int q = 0; q < E_; ++q) nodes[(size_t)idx * E_ + q] = e[q];
        m = valid[s] ? 1.0f : 0.0f;
    } else {
        int k = j - 2;
        for (int q = 0; q < E_; ++q) nodes[(size_t)idx * E_ + q] = lew[k * E_ + q] + leb[q];
        m = (valid[s] && lact[s * 7 + k]) ? 1.0f : 0.0f;
    }
    nm[idx] = m;
}

__global__ void k_edges(const float* __restrict__ nm, const int* __restrict__ valid,
        const int* __restrict__ nxt, int* __restrict__ esrc, int* __restrict__ edst,
        float* __restrict__ ew) {
    int e = blockIdx.x * blockDim.x + threadIdx.x;
    if (e >= NE_) return;
    if (e < NEPAT_) {
        int s = e / 58, p = e % 58;
        int ls = c_srcl[p], ld = c_dstl[p];
        esrc[e] = s * 9 + ls;
        edst[e] = s * 9 + ld;
        ew[e] = nm[s * 9 + ls] * nm[s * 9 + ld];
    } else {
        int r = e - NEPAT_;
        int s = (r < S_) ? r : (r - S_);
        int cs = s * 9;
        int cd = ((nxt[s] > 0) ? nxt[s] : 0) * 9;
        float cw = (valid[s] && nxt[s] >= 0) ? 1.0f : 0.0f;
        if (r < S_) { esrc[e] = cs; edst[e] = cd; }
        else        { esrc[e] = cd; edst[e] = cs; }
        ew[e] = cw;
    }
}

__global__ void k_deg_acc(const int* __restrict__ edst, const float* __restrict__ ew,
                          float* __restrict__ deg, int ne) {
    int e = blockIdx.x * blockDim.x + threadIdx.x;
    if (e < ne) atomicAdd(&deg[edst[e]], ew[e]);
}
__global__ void k_dinv(const float* __restrict__ deg, float* __restrict__ dinv, int n) {
    int i = blockIdx.x * blockDim.x + threadIdx.x;
    if (i < n) dinv[i] = rsqrtf(deg[i]);
}
__global__ void k_gcn_mm(const float* __restrict__ hin, const float* __restrict__ W,
                         float* __restrict__ hout, int n, int ein, int eout) {
    int idx = blockIdx.x * blockDim.x + threadIdx.x;
    if (idx >= n * eout) return;
    int node = idx / eout, o = idx % eout;
    float a = 0.0f;
    for (int k = 0; k < ein; ++k) a += hin[(size_t)node * ein + k] * W[k * eout + o];
    hout[idx] = a;
}
__global__ void k_gcn_agg(const int* __restrict__ esrc, const int* __restrict__ edst,
        const float* __restrict__ ew, const float* __restrict__ dinv,
        const float* __restrict__ hlin, float* __restrict__ agg, int ne, int eout) {
    int idx = blockIdx.x * blockDim.x + threadIdx.x;
    if (idx >= ne * eout) return;
    int e = idx / eout, o = idx % eout;
    int sN = esrc[e], dN = edst[e];
    float w = dinv[sN] * dinv[dN] * ew[e];
    atomicAdd(&agg[(size_t)dN * eout + o], w * hlin[(size_t)sN * eout + o]);
}
__global__ void k_gcn_fin(const float* __restrict__ agg, const float* __restrict__ hlin,
        const float* __restrict__ deg, const float* __restrict__ bias,
        float* __restrict__ hout, int n, int eout) {
    int idx = blockIdx.x * blockDim.x + threadIdx.x;
    if (idx >= n * eout) return;
    int node = idx / eout, o = idx % eout;
    float v = agg[idx] + hlin[idx] / deg[node] + bias[o];
    hout[idx] = fmaxf(v, 0.0f);
}

__global__ void k_samp(const float* __restrict__ h2, const float* __restrict__ nm,
                       float* __restrict__ samp) {
    int idx = blockIdx.x * blockDim.x + threadIdx.x;
    if (idx >= S_ * GOUT_) return;
    int s = idx / GOUT_, o = idx % GOUT_;
    float cnt = 0.0f, acc = 0.0f;
    for (int j = 0; j < 9; ++j) {
        float m = nm[s * 9 + j];
        cnt += m;
        acc += m * h2[(size_t)(s * 9 + j) * GOUT_ + o];
    }
    samp[idx] = acc / fmaxf(cnt, 1.0f);
}
__global__ void k_gpool(const float* __restrict__ samp, const int* __restrict__ valid,
                        float* __restrict__ gpool) {
    int idx = blockIdx.x * blockDim.x + threadIdx.x;
    if (idx >= B_ * GOUT_) return;
    int b = idx / GOUT_, o = idx % GOUT_;
    float den = 0.0f, num = 0.0f;
    for (int t = 0; t < T_; ++t) {
        float v = valid[b * T_ + t] ? 1.0f : 0.0f;
        den += v;
        num += v * samp[(size_t)(b * T_ + t) * GOUT_ + o];
    }
    gpool[idx] = num / fmaxf(den, 1.0f);
}

__global__ __launch_bounds__(256) void k_diag(const float* __restrict__ pooled,
        const float* __restrict__ gpool, const float* __restrict__ wd,
        const float* __restrict__ bd, float* __restrict__ outD,
        float* __restrict__ outC, float* __restrict__ outE) {
    __shared__ float sh[256];
    int b = blockIdx.x, tid = threadIdx.x;
    float pz[7] = {};
    for (int d = tid; d < D_ + GOUT_; d += 256) {
        float c = (d < D_) ? pooled[(size_t)b * D_ + d] : gpool[b * GOUT_ + (d - D_)];
        #pragma unroll
        for (int j = 0; j < 7; ++j) pz[j] += c * wd[d * 7 + j];
    }
    float z[7];
    for (int j = 0; j < 7; ++j) z[j] = block_reduce_sum(pz[j], sh) + bd[j];
    if (tid == 0) {
        float mx = z[0];
        for (int j = 0; j < 7; ++j) { outD[b * 7 + j] = z[j]; mx = fmaxf(mx, z[j]); }
        float e[7], sum = 0.0f;
        for (int j = 0; j < 7; ++j) { e[j] = expf(z[j] - mx); sum += e[j]; }
        float conf = 0.0f, ent = 0.0f;
        for (int j = 0; j < 7; ++j) {
            float pr = e[j] / sum;
            conf = fmaxf(conf, pr);
            ent -= pr * logf(pr + 1e-8f);
        }
        outC[b] = conf;
        outE[b] = ent / logf(7.0f);
    }
}

// ============================ host launch ============================
extern "C" void kernel_launch(void* const* d_in, const int* in_sizes, int n_in,
                              void* d_out, int out_size, void* d_ws, size_t ws_size,
                              hipStream_t stream) {
    const float* feat      = (const float*)d_in[0];
    const float* w_phase   = (const float*)d_in[1];
    const float* b_phase   = (const float*)d_in[2];
    const float* w_loc     = (const float*)d_in[3];
    const float* b_loc     = (const float*)d_in[4];
    const float* w_les     = (const float*)d_in[5];
    const float* b_les     = (const float*)d_in[6];
    const float* emb_phase = (const float*)d_in[7];
    const float* emb_loc   = (const float*)d_in[8];
    const float* les_emb_w = (const float*)d_in[9];
    const float* les_emb_b = (const float*)d_in[10];
    const float* qkv_w     = (const float*)d_in[11];
    const float* qkv_b     = (const float*)d_in[12];
    const float* o_w       = (const float*)d_in[13];
    const float* o_b       = (const float*)d_in[14];
    const float* ln1_g     = (const float*)d_in[15];
    const float* ln1_b     = (const float*)d_in[16];
    const float* ln2_g     = (const float*)d_in[17];
    const float* ln2_b     = (const float*)d_in[18];
    const float* ff1_w     = (const float*)d_in[19];
    const float* ff1_b     = (const float*)d_in[20];
    const float* ff2_w     = (const float*)d_in[21];
    const float* ff2_b     = (const float*)d_in[22];
    const float* w_score   = (const float*)d_in[23];
    const float* b_score   = (const float*)d_in[24];
    const float* gcn1_w    = (const float*)d_in[25];
    const float* gcn1_b    = (const float*)d_in[26];
    const float* gcn2_w    = (const float*)d_in[27];
    const float* gcn2_b    = (const float*)d_in[28];
    const float* w_diag    = (const float*)d_in[29];
    const float* b_diag    = (const float*)d_in[30];
    const int*   mask      = (const int*)d_in[31];

    float* out = (float*)d_out;
    float* outPhase = out;
    float* outLoc   = out + S_ * 3;
    float* outLes   = out + S_ * 3 + S_ * 7;
    float* outDiag  = out + S_ * 3 + 2 * S_ * 7;
    float* outConf  = outDiag + B_ * 7;
    float* outEnt   = outConf + B_;

    char* wp = (char*)d_ws;
    auto alloc = [&](size_t bytes) -> void* {
        void* r = (void*)wp;
        wp += (bytes + 255) & ~(size_t)255;
        return r;
    };
    float* x      = (float*)alloc(sizeof(float) * S_ * D_);
    float* tmp    = (float*)alloc(sizeof(float) * S_ * D_);
    float* qkv    = (float*)alloc(sizeof(float) * S_ * 3 * D_);
    float* buf1   = (float*)alloc(sizeof(float) * S_ * DFF_);        // ctx / ff1 out
    u16*   act_bf = (u16*)  alloc(sizeof(u16) * S_ * DFF_);
    u16*   wbuf   = (u16*)  alloc(sizeof(u16) * (size_t)3 * D_ * D_);
    float* scores = (float*)alloc(sizeof(float) * S_);
    float* wts    = (float*)alloc(sizeof(float) * S_);
    int*   pid    = (int*)  alloc(sizeof(int) * S_);
    int*   lid    = (int*)  alloc(sizeof(int) * S_);
    int*   lact   = (int*)  alloc(sizeof(int) * S_ * 7);
    int*   valid  = (int*)  alloc(sizeof(int) * S_);
    int*   nxt    = (int*)  alloc(sizeof(int) * S_);
    float* pooled = (float*)alloc(sizeof(float) * B_ * D_);
    float* nodes  = (float*)alloc(sizeof(float) * N9_ * E_);
    float* nm     = (float*)alloc(sizeof(float) * S_ * 9);
    int*   esrc   = (int*)  alloc(sizeof(int) * NE_);
    int*   edst   = (int*)  alloc(sizeof(int) * NE_);
    float* eww    = (float*)alloc(sizeof(float) * NE_);
    float* deg    = (float*)alloc(sizeof(float) * N9_);
    float* dinv   = (float*)alloc(sizeof(float) * N9_);
    float* hlin   = (float*)alloc(sizeof(float) * N9_ * HID_);
    float* agg    = (float*)alloc(sizeof(float) * N9_ * HID_);
    float* h1     = (float*)alloc(sizeof(float) * N9_ * HID_);
    float* hlin2  = (float*)alloc(sizeof(float) * N9_ * GOUT_);
    float* h2     = (float*)alloc(sizeof(float) * N9_ * GOUT_);
    float* samp   = (float*)alloc(sizeof(float) * S_ * GOUT_);
    float* gpool  = (float*)alloc(sizeof(float) * B_ * GOUT_);

    auto tobf = [&](const float* src, u16* dst, size_t n) {
        k_f32_to_bf16<<<(unsigned)((n + 255) / 256), 256, 0, stream>>>(src, dst, (int)n);
    };
    auto gemm = [&](const u16* A, const u16* W, const float* bias, float* C,
                    int M, int N, int K, int relu) {
        dim3 g((unsigned)(N / 128), (unsigned)(M / 64));
        k_gemm<<<g, 128, 0, stream>>>(A, W, bias, C, M, N, K, relu);
    };

    // Heads straight off feat (also writes argmax ids / lesion activations)
    k_heads<<<S_, 256, 0, stream>>>(feat, w_phase, b_phase, w_loc, b_loc, w_les, b_les,
                                    outPhase, outLoc, outLes, pid, lid, lact);

    // x = feat
    k_copy<<<(S_ * D_ + 255) / 256, 256, 0, stream>>>(feat, x, S_ * D_);

    for (int l = 0; l < L_; ++l) {
        // QKV projection (WMMA bf16)
        tobf(x, act_bf, (size_t)S_ * D_);
        tobf(qkv_w + (size_t)l * 3 * D_ * D_, wbuf, (size_t)3 * D_ * D_);
        gemm(act_bf, wbuf, qkv_b + (size_t)l * 3 * D_, qkv, S_, 3 * D_, D_, 0);
        // Attention
        k_attn<<<B_ * H_, 256, 0, stream>>>(qkv, mask, buf1);
        // Output projection
        tobf(buf1, act_bf, (size_t)S_ * D_);
        tobf(o_w + (size_t)l * D_ * D_, wbuf, (size_t)D_ * D_);
        gemm(act_bf, wbuf, o_b + (size_t)l * D_, tmp, S_, D_, D_, 0);
        // x = LN(x + proj)
        k_resln<<<S_, 256, 0, stream>>>(x, tmp, ln1_g + (size_t)l * D_, ln1_b + (size_t)l * D_, x);
        // FF1 (relu fused)
        tobf(x, act_bf, (size_t)S_ * D_);
        tobf(ff1_w + (size_t)l * DFF_ * D_, wbuf, (size_t)DFF_ * D_);
        gemm(act_bf, wbuf, ff1_b + (size_t)l * DFF_, buf1, S_, DFF_, D_, 1);
        // FF2
        tobf(buf1, act_bf, (size_t)S_ * DFF_);
        tobf(ff2_w + (size_t)l * D_ * DFF_, wbuf, (size_t)D_ * DFF_);
        gemm(act_bf, wbuf, ff2_b + (size_t)l * D_, tmp, S_, D_, DFF_, 0);
        // x = LN(x + ff)
        k_resln<<<S_, 256, 0, stream>>>(x, tmp, ln2_g + (size_t)l * D_, ln2_b + (size_t)l * D_, x);
    }

    // Attention pooling over T with mask
    k_score<<<S_, 256, 0, stream>>>(x, w_score, b_score, scores);
    k_wts<<<1, 64, 0, stream>>>(scores, mask, wts, valid);
    k_pooled<<<B_, 256, 0, stream>>>(x, wts, pooled);

    // GNN graph build
    k_scan<<<1, 1, 0, stream>>>(valid, nxt);
    k_nodes<<<(N9_ + 255) / 256, 256, 0, stream>>>(emb_phase, emb_loc, les_emb_w, les_emb_b,
                                                   pid, lid, lact, valid, nodes, nm);
    k_edges<<<(NE_ + 255) / 256, 256, 0, stream>>>(nm, valid, nxt, esrc, edst, eww);
    k_fill<<<(N9_ + 255) / 256, 256, 0, stream>>>(deg, 1.0f, N9_);
    k_deg_acc<<<(NE_ + 255) / 256, 256, 0, stream>>>(edst, eww, deg, NE_);
    k_dinv<<<(N9_ + 255) / 256, 256, 0, stream>>>(deg, dinv, N9_);

    // GCN layer 1: E_ -> HID_
    k_gcn_mm<<<(N9_ * HID_ + 255) / 256, 256, 0, stream>>>(nodes, gcn1_w, hlin, N9_, E_, HID_);
    k_fill<<<(N9_ * HID_ + 255) / 256, 256, 0, stream>>>(agg, 0.0f, N9_ * HID_);
    k_gcn_agg<<<(NE_ * HID_ + 255) / 256, 256, 0, stream>>>(esrc, edst, eww, dinv, hlin, agg, NE_, HID_);
    k_gcn_fin<<<(N9_ * HID_ + 255) / 256, 256, 0, stream>>>(agg, hlin, deg, gcn1_b, h1, N9_, HID_);

    // GCN layer 2: HID_ -> GOUT_
    k_gcn_mm<<<(N9_ * GOUT_ + 255) / 256, 256, 0, stream>>>(h1, gcn2_w, hlin2, N9_, HID_, GOUT_);
    k_fill<<<(N9_ * GOUT_ + 255) / 256, 256, 0, stream>>>(agg, 0.0f, N9_ * GOUT_);
    k_gcn_agg<<<(NE_ * GOUT_ + 255) / 256, 256, 0, stream>>>(esrc, edst, eww, dinv, hlin2, agg, NE_, GOUT_);
    k_gcn_fin<<<(N9_ * GOUT_ + 255) / 256, 256, 0, stream>>>(agg, hlin2, deg, gcn2_b, h2, N9_, GOUT_);

    // Pool GNN + diagnosis head
    k_samp<<<(S_ * GOUT_ + 255) / 256, 256, 0, stream>>>(h2, nm, samp);
    k_gpool<<<(B_ * GOUT_ + 255) / 256, 256, 0, stream>>>(samp, valid, gpool);
    k_diag<<<B_, 256, 0, stream>>>(pooled, gpool, w_diag, b_diag, outDiag, outConf, outEnt);
}
